// CEABlock_43559558316834
// MI455X (gfx1250) — compile-verified, matrix operand order checked by reading, not run on hardware
//
#include <hip/hip_runtime.h>
#include <hip/hip_bf16.h>
#include <math.h>

// ---------------- problem constants ----------------
#define BB   32
#define LLQ  320
#define CCH  768
#define NH   12
#define DHD  64
#define LT   64
#define LS   256      // spatial tokens
#define LKP  180      // ceil(0.7*256)
#define LRM  76
#define LP   244      // 64 + 180
#define ADH  64
#define FF   3072

typedef __attribute__((ext_vector_type(16))) __bf16       v16bf;
typedef __attribute__((ext_vector_type(2)))  __bf16       v2bf;
typedef __attribute__((ext_vector_type(8)))  float        v8f;
typedef __attribute__((ext_vector_type(8)))  unsigned int v8u;

// f32 -> packed bf16 pair using native fptrunc (RNE); lets the backend form
// v_cvt_pk_bf16_f32 instead of an emulated integer-rounding sequence.
__device__ __forceinline__ unsigned int pack_bf2(float lo, float hi) {
  v2bf v;
  v[0] = (__bf16)lo;
  v[1] = (__bf16)hi;
  return __builtin_bit_cast(unsigned int, v);
}
__device__ __forceinline__ float gelu_exact(float x) {
  return 0.5f * x * (1.0f + erff(x * 0.7071067811865475f));
}

// ---------------------------------------------------------------------------
// Batched GEMM:  C = act( alpha * A @ B(^T) + bias )
// Requirements (all satisfied by this workload): M%64==0, N%64==0, K%32==0,
// all base pointers / leading dims 16-byte aligned.
// 64x64 macro tile, 128 threads (4 waves, wave32), each wave a 2x2 grid of
// v_wmma_f32_16x16x32_bf16.  Double-buffered LDS with register prefetch.
// Batch z: z1=z/sub, z2=z%sub; pointers offset by z1*S1+z2*S2 (covers
// per-(b,h) attention addressing).
// ---------------------------------------------------------------------------
#define TM 64
#define TN 64
#define TK 32

template<int TRANSB, int ACT>
__global__ void __launch_bounds__(128)
gemm_bf16_wmma_kernel(const float* __restrict__ A, const float* __restrict__ Bm,
                      const float* __restrict__ bias, float* __restrict__ Cm,
                      int K, int lda, int ldb, int ldc,
                      long long aS1, long long aS2,
                      long long bS1, long long bS2,
                      long long cS1, long long cS2,
                      int sub, float alpha)
{
  const int z  = blockIdx.z;
  const int z1 = z / sub;
  const int z2 = z - z1 * sub;
  A  += (long long)z1 * aS1 + (long long)z2 * aS2;
  Bm += (long long)z1 * bS1 + (long long)z2 * bS2;
  Cm += (long long)z1 * cS1 + (long long)z2 * cS2;

  __shared__ __align__(16) unsigned short lA[2][TM][TK + 8];
  __shared__ __align__(16) unsigned short lB[2][TN][TK + 8];  // [n][k] layout

  const int tid  = threadIdx.x;
  const int wave = tid >> 5;
  const int lane = tid & 31;
  const int wm   = (wave >> 1) * 32;
  const int wn   = (wave & 1) * 32;
  const int hl   = lane >> 4;       // lane half per ISA frag layout
  const int l16  = lane & 15;

  const int m0 = blockIdx.y * TM;
  const int n0 = blockIdx.x * TN;

  // staging coords (128 threads, 4 float4 each per matrix per tile)
  const int sar  = tid >> 3;   // A row base (+16/pass), 8 float4 per 32-wide row
  const int sac4 = tid & 7;
  const int sbkr = tid >> 4;   // B(non-T) k-row base (+8/pass), 16 float4 per 64-wide row
  const int sbc4 = tid & 15;

  float4 pa[4], pb[4];

  auto stageLoad = [&](int kt) {
    const int kk = kt * TK;
#pragma unroll
    for (int p = 0; p < 4; ++p) {
      pa[p] = *(const float4*)(A + (long long)(m0 + sar + p * 16) * lda + kk + sac4 * 4);
      if (TRANSB)
        pb[p] = *(const float4*)(Bm + (long long)(n0 + sar + p * 16) * ldb + kk + sac4 * 4);
      else
        pb[p] = *(const float4*)(Bm + (long long)(kk + sbkr + p * 8) * ldb + n0 + sbc4 * 4);
    }
  };
  auto stageStore = [&](int buf) {
#pragma unroll
    for (int p = 0; p < 4; ++p) {
      uint2 ua; ua.x = pack_bf2(pa[p].x, pa[p].y); ua.y = pack_bf2(pa[p].z, pa[p].w);
      *(uint2*)&lA[buf][sar + p * 16][sac4 * 4] = ua;
      if (TRANSB) {
        uint2 ub; ub.x = pack_bf2(pb[p].x, pb[p].y); ub.y = pack_bf2(pb[p].z, pb[p].w);
        *(uint2*)&lB[buf][sar + p * 16][sac4 * 4] = ub;
      } else {
        const int kr = sbkr + p * 8;
        unsigned int p01 = pack_bf2(pb[p].x, pb[p].y);
        unsigned int p23 = pack_bf2(pb[p].z, pb[p].w);
        lB[buf][sbc4 * 4 + 0][kr] = (unsigned short)p01;
        lB[buf][sbc4 * 4 + 1][kr] = (unsigned short)(p01 >> 16);
        lB[buf][sbc4 * 4 + 2][kr] = (unsigned short)p23;
        lB[buf][sbc4 * 4 + 3][kr] = (unsigned short)(p23 >> 16);
      }
    }
  };

  v8f acc00 = {}, acc01 = {}, acc10 = {}, acc11 = {};

  const int nk = K / TK;
  stageLoad(0);
  stageStore(0);
  __syncthreads();

  for (int kt = 0; kt < nk; ++kt) {
    const int cur = kt & 1;
    if (kt + 1 < nk) stageLoad(kt + 1);   // prefetch next tile into registers

    // fragments per ISA 16-bit 16x32 layout: half hl picks k-groups
    v8u au0, au1, bu0, bu1;
    const int kb = hl * 8;
#pragma unroll
    for (int q = 0; q < 8; ++q) {
      int kq = ((q & 4) ? 16 : 0) + kb + ((q & 3) << 1);
      au0[q] = *(const unsigned int*)&lA[cur][wm + l16][kq];
      au1[q] = *(const unsigned int*)&lA[cur][wm + 16 + l16][kq];
      bu0[q] = *(const unsigned int*)&lB[cur][wn + l16][kq];
      bu1[q] = *(const unsigned int*)&lB[cur][wn + 16 + l16][kq];
    }
    v16bf a0 = __builtin_bit_cast(v16bf, au0);
    v16bf a1 = __builtin_bit_cast(v16bf, au1);
    v16bf b0 = __builtin_bit_cast(v16bf, bu0);
    v16bf b1 = __builtin_bit_cast(v16bf, bu1);

    acc00 = __builtin_amdgcn_wmma_f32_16x16x32_bf16(false, a0, false, b0, (short)0, acc00, false, false);
    acc01 = __builtin_amdgcn_wmma_f32_16x16x32_bf16(false, a0, false, b1, (short)0, acc01, false, false);
    acc10 = __builtin_amdgcn_wmma_f32_16x16x32_bf16(false, a1, false, b0, (short)0, acc10, false, false);
    acc11 = __builtin_amdgcn_wmma_f32_16x16x32_bf16(false, a1, false, b1, (short)0, acc11, false, false);

    if (kt + 1 < nk) stageStore(cur ^ 1);
    __syncthreads();
  }

  // epilogue (C layout: vgpr e, lanes0-15 -> M=e, lanes16-31 -> M=e+8)
#pragma unroll
  for (int ti = 0; ti < 2; ++ti) {
#pragma unroll
    for (int tj = 0; tj < 2; ++tj) {
      v8f acc = (ti == 0) ? (tj == 0 ? acc00 : acc01)
                          : (tj == 0 ? acc10 : acc11);
      const int col = n0 + wn + tj * 16 + l16;
      float bv = bias ? bias[col] : 0.0f;
#pragma unroll
      for (int e = 0; e < 8; ++e) {
        int row = m0 + wm + ti * 16 + hl * 8 + e;
        float o = acc[e] * alpha + bv;
        if (ACT) o = gelu_exact(o);
        Cm[(long long)row * ldc + col] = o;
      }
    }
  }
}

// ---------------- LayerNorm (one block per row) ----------------
__global__ void __launch_bounds__(256)
ln_kernel(const float* __restrict__ xin, const float* __restrict__ g,
          const float* __restrict__ bta, float* __restrict__ out, int C)
{
  long long row = blockIdx.x;
  const float* xr = xin + row * C;
  float* orow = out + row * C;
  __shared__ float red[256];
  int tid = threadIdx.x;

  float s = 0.0f;
  for (int i = tid; i < C; i += 256) s += xr[i];
  red[tid] = s; __syncthreads();
  for (int st = 128; st > 0; st >>= 1) { if (tid < st) red[tid] += red[tid + st]; __syncthreads(); }
  float mean = red[0] / (float)C; __syncthreads();

  float v = 0.0f;
  for (int i = tid; i < C; i += 256) { float d = xr[i] - mean; v += d * d; }
  red[tid] = v; __syncthreads();
  for (int st = 128; st > 0; st >>= 1) { if (tid < st) red[tid] += red[tid + st]; __syncthreads(); }
  float inv = rsqrtf(red[0] / (float)C + 1e-5f);

  for (int i = tid; i < C; i += 256)
    orow[i] = (xr[i] - mean) * inv * g[i] + bta[i];
}

// ---------------- row softmax, in place ----------------
__global__ void __launch_bounds__(128)
softmax_kernel(float* __restrict__ data, int n)
{
  long long row = blockIdx.x;
  float* d = data + row * (long long)n;
  __shared__ float red[128];
  int tid = threadIdx.x;

  float m = -1e30f;
  for (int i = tid; i < n; i += 128) m = fmaxf(m, d[i]);
  red[tid] = m; __syncthreads();
  for (int st = 64; st > 0; st >>= 1) { if (tid < st) red[tid] = fmaxf(red[tid], red[tid + st]); __syncthreads(); }
  m = red[0]; __syncthreads();

  float s = 0.0f;
  for (int i = tid; i < n; i += 128) { float e = __expf(d[i] - m); d[i] = e; s += e; }
  red[tid] = s; __syncthreads();
  for (int st = 64; st > 0; st >>= 1) { if (tid < st) red[tid] += red[tid + st]; __syncthreads(); }
  float inv = 1.0f / red[0];

  for (int i = tid; i < n; i += 128) d[i] *= inv;
}

// ---------------- mean over heads & first 64 query rows ----------------
__global__ void attn_mean_kernel(const float* __restrict__ attn, float* __restrict__ attn_t)
{
  int b = blockIdx.x;
  for (int j = threadIdx.x; j < LS; j += blockDim.x) {
    float s = 0.0f;
    for (int h = 0; h < NH; ++h)
      for (int i = 0; i < LT; ++i)
        s += attn[(((long long)(b * NH + h) * LLQ) + i) * LLQ + LT + j];
    attn_t[b * LS + j] = s * (1.0f / (NH * LT));
  }
}

// ---------------- stable descending argsort via rank counting ----------------
__global__ void __launch_bounds__(256)
rank_kernel(const float* __restrict__ attn_t, int* __restrict__ order)
{
  int b = blockIdx.x, j = threadIdx.x;
  __shared__ float v[LS];
  v[j] = attn_t[b * LS + j];
  __syncthreads();
  float mv = v[j];
  int r = 0;
  for (int t = 0; t < LS; ++t)
    r += (v[t] > mv) || (v[t] == mv && t < j);
  order[b * LS + r] = j;
}

// ---------------- keep / removed index outputs (as float) ----------------
__global__ void __launch_bounds__(256)
ce_indices_kernel(const int* __restrict__ order, const int* __restrict__ gidx,
                  float* __restrict__ keep, float* __restrict__ removed)
{
  int b = blockIdx.x, t = threadIdx.x;
  int o = order[b * LS + t];
  float val = (float)gidx[b * LS + o];
  if (t < LKP) keep[b * LKP + t] = val;
  else         removed[b * LRM + (t - LKP)] = val;
}

__global__ void copy_i2f_kernel(const int* __restrict__ in, float* __restrict__ out, long long n)
{
  long long i = (long long)blockIdx.x * blockDim.x + threadIdx.x;
  if (i < n) out[i] = (float)in[i];
}

// gather pruned tokens: rows 0..63 pass through, rows 64.. gathered by order
__global__ void gather_kernel(const float* __restrict__ xin, const int* __restrict__ order,
                              float* __restrict__ out)
{
  long long i = (long long)blockIdx.x * blockDim.x + threadIdx.x;
  const long long total = (long long)BB * LP * CCH;
  if (i >= total) return;
  int c = (int)(i % CCH);
  long long r = i / CCH;
  int row = (int)(r % LP);
  int b   = (int)(r / LP);
  int src = (row < LT) ? row : (LT + order[b * LS + (row - LT)]);
  out[i] = xin[((long long)b * LLQ + src) * CCH + c];
}

__global__ void add3_kernel(const float* __restrict__ a, const float* __restrict__ b,
                            const float* __restrict__ c, float* __restrict__ o, long long n)
{
  long long i = (long long)blockIdx.x * blockDim.x + threadIdx.x;
  if (i < n) o[i] = a[i] + b[i] + c[i];
}

__global__ void addip2_kernel(float* __restrict__ o, const float* __restrict__ a,
                              const float* __restrict__ b, long long n)
{
  long long i = (long long)blockIdx.x * blockDim.x + threadIdx.x;
  if (i < n) o[i] = o[i] + a[i] + b[i];
}

// ---------------------------------------------------------------------------
// host side
// ---------------------------------------------------------------------------
static inline void launch_gemm(hipStream_t st, const float* A, const float* Bm,
                               const float* bias, float* Cm,
                               int M, int N, int K, int lda, int ldb, int ldc,
                               long long aS1, long long aS2, long long bS1, long long bS2,
                               long long cS1, long long cS2,
                               int batch, int sub, float alpha, int transB, int act)
{
  dim3 g((unsigned)(N / TN), (unsigned)(M / TM), (unsigned)batch);
  if (transB)
    gemm_bf16_wmma_kernel<1, 0><<<g, 128, 0, st>>>(A, Bm, bias, Cm, K, lda, ldb, ldc,
                                                   aS1, aS2, bS1, bS2, cS1, cS2, sub, alpha);
  else if (act)
    gemm_bf16_wmma_kernel<0, 1><<<g, 128, 0, st>>>(A, Bm, bias, Cm, K, lda, ldb, ldc,
                                                   aS1, aS2, bS1, bS2, cS1, cS2, sub, alpha);
  else
    gemm_bf16_wmma_kernel<0, 0><<<g, 128, 0, st>>>(A, Bm, bias, Cm, K, lda, ldb, ldc,
                                                   aS1, aS2, bS1, bS2, cS1, cS2, sub, alpha);
}

extern "C" void kernel_launch(void* const* d_in, const int* in_sizes, int n_in,
                              void* d_out, int out_size, void* d_ws, size_t ws_size,
                              hipStream_t stream)
{
  (void)in_sizes; (void)n_in; (void)out_size; (void)ws_size;

  const float* x      = (const float*)d_in[0];
  const float* xi     = (const float*)d_in[1];
  const int*   gi_t   = (const int*)d_in[2];
  const int*   gi_ti  = (const int*)d_in[3];
  const int*   gi_s   = (const int*)d_in[4];
  const int*   gi_si  = (const int*)d_in[5];
  const float* n1_g   = (const float*)d_in[7];
  const float* n1_b   = (const float*)d_in[8];
  const float* qkv_w  = (const float*)d_in[9];
  const float* proj_w = (const float*)d_in[10];
  const float* proj_b = (const float*)d_in[11];
  const float* n2_g   = (const float*)d_in[12];
  const float* n2_b   = (const float*)d_in[13];
  const float* fc1_w  = (const float*)d_in[14];
  const float* fc1_b  = (const float*)d_in[15];
  const float* fc2_w  = (const float*)d_in[16];
  const float* fc2_b  = (const float*)d_in[17];
  const float* ad1_dw = (const float*)d_in[18];
  const float* ad1_db = (const float*)d_in[19];
  const float* ad1_uw = (const float*)d_in[20];
  const float* ad1_ub = (const float*)d_in[21];
  const float* ad2_dw = (const float*)d_in[22];
  const float* ad2_db = (const float*)d_in[23];
  const float* ad2_uw = (const float*)d_in[24];
  const float* ad2_ub = (const float*)d_in[25];

  // output layout (floats, in reference return order)
  const size_t X_SZ  = (size_t)BB * LP * CCH;
  const size_t GT_SZ = (size_t)BB * LT;
  const size_t GS_SZ = (size_t)BB * LKP;
  const size_t RM_SZ = (size_t)BB * LRM;
  const size_t AT_SZ = (size_t)BB * NH * LLQ * LLQ;
  float* out = (float*)d_out;
  float* o_x    = out;
  float* o_gt   = o_x    + X_SZ;
  float* o_gs   = o_gt   + GT_SZ;
  float* o_rm   = o_gs   + GS_SZ;
  float* o_attn = o_rm   + RM_SZ;
  float* o_xi   = o_attn + AT_SZ;
  float* o_gti  = o_xi   + X_SZ;
  float* o_gsi  = o_gti  + GT_SZ;
  float* o_rmi  = o_gsi  + GS_SZ;
  float* o_iat  = o_rmi  + RM_SZ;

  // workspace layout (floats)
  const size_t BLC = (size_t)BB * LLQ * CCH;       // 7,864,320
  const size_t BIG = (size_t)BB * LP * FF;         // 23,986,176 (>= B*L*3C)
  float* w    = (float*)d_ws;
  float* lnA  = w;
  float* lnB  = lnA + BLC;
  float* big  = lnB + BLC;                          // qkv / MLP hidden
  float* t1   = big + BIG;
  float* t2   = t1  + BLC;
  float* mh   = t2  + BLC;                          // adapter hidden (B*L*64)
  float* xn   = mh  + (size_t)BB * LLQ * ADH;
  float* xin  = xn  + BLC;
  float* at   = xin + BLC;                          // attn_t (x), B*256
  float* ati  = at  + (size_t)BB * LS;
  int*   ord  = (int*)(ati + (size_t)BB * LS);
  int*   ordi = ord + (size_t)BB * LS;

  const long long L3C = (long long)LLQ * 3 * CCH;  // per-batch qkv stride
  const long long HLL = (long long)NH * LLQ * LLQ;
  const long long LLl = (long long)LLQ * LLQ;

  // ======================= stage 1: LN =======================
  ln_kernel<<<BB * LLQ, 256, 0, stream>>>(x,  n1_g, n1_b, lnA, CCH);
  ln_kernel<<<BB * LLQ, 256, 0, stream>>>(xi, n1_g, n1_b, lnB, CCH);

  // ---------- x branch attention ----------
  launch_gemm(stream, lnA, qkv_w, nullptr, big,
              BB * LLQ, 3 * CCH, CCH, CCH, 3 * CCH, 3 * CCH,
              0, 0, 0, 0, 0, 0, 1, 1, 1.0f, 0, 0);
  launch_gemm(stream, big, big + CCH, nullptr, o_attn,           // S = (Q/8) K^T
              LLQ, LLQ, DHD, 3 * CCH, 3 * CCH, LLQ,
              L3C, DHD, L3C, DHD, HLL, LLl,
              BB * NH, NH, 0.125f, 1, 0);
  softmax_kernel<<<BB * NH * LLQ, 128, 0, stream>>>(o_attn, LLQ);
  launch_gemm(stream, o_attn, big + 2 * CCH, nullptr, t1,        // O = P V
              LLQ, DHD, LLQ, LLQ, 3 * CCH, CCH,
              HLL, LLl, L3C, DHD, (long long)LLQ * CCH, DHD,
              BB * NH, NH, 1.0f, 0, 0);
  launch_gemm(stream, t1, proj_w, proj_b, t2,                    // xa
              BB * LLQ, CCH, CCH, CCH, CCH, CCH,
              0, 0, 0, 0, 0, 0, 1, 1, 1.0f, 0, 0);
  // mixer1 on ln(xi)
  launch_gemm(stream, lnB, ad1_dw, ad1_db, mh,
              BB * LLQ, ADH, CCH, CCH, ADH, ADH,
              0, 0, 0, 0, 0, 0, 1, 1, 1.0f, 0, 1);
  launch_gemm(stream, mh, ad1_uw, ad1_ub, t1,
              BB * LLQ, CCH, ADH, ADH, CCH, CCH,
              0, 0, 0, 0, 0, 0, 1, 1, 1.0f, 0, 0);
  add3_kernel<<<(unsigned)((BLC + 255) / 256), 256, 0, stream>>>(x, t2, t1, xn, (long long)BLC);

  // ---------- xi branch attention ----------
  launch_gemm(stream, lnB, qkv_w, nullptr, big,
              BB * LLQ, 3 * CCH, CCH, CCH, 3 * CCH, 3 * CCH,
              0, 0, 0, 0, 0, 0, 1, 1, 1.0f, 0, 0);
  launch_gemm(stream, big, big + CCH, nullptr, o_iat,
              LLQ, LLQ, DHD, 3 * CCH, 3 * CCH, LLQ,
              L3C, DHD, L3C, DHD, HLL, LLl,
              BB * NH, NH, 0.125f, 1, 0);
  softmax_kernel<<<BB * NH * LLQ, 128, 0, stream>>>(o_iat, LLQ);
  launch_gemm(stream, o_iat, big + 2 * CCH, nullptr, t1,
              LLQ, DHD, LLQ, LLQ, 3 * CCH, CCH,
              HLL, LLl, L3C, DHD, (long long)LLQ * CCH, DHD,
              BB * NH, NH, 1.0f, 0, 0);
  launch_gemm(stream, t1, proj_w, proj_b, t2,
              BB * LLQ, CCH, CCH, CCH, CCH, CCH,
              0, 0, 0, 0, 0, 0, 1, 1, 1.0f, 0, 0);
  // mixer1 on ln(xori) = ln(x)
  launch_gemm(stream, lnA, ad1_dw, ad1_db, mh,
              BB * LLQ, ADH, CCH, CCH, ADH, ADH,
              0, 0, 0, 0, 0, 0, 1, 1, 1.0f, 0, 1);
  launch_gemm(stream, mh, ad1_uw, ad1_ub, t1,
              BB * LLQ, CCH, ADH, ADH, CCH, CCH,
              0, 0, 0, 0, 0, 0, 1, 1, 1.0f, 0, 0);
  add3_kernel<<<(unsigned)((BLC + 255) / 256), 256, 0, stream>>>(xi, t2, t1, xin, (long long)BLC);

  // ======================= CE token pruning =======================
  attn_mean_kernel<<<BB, 256, 0, stream>>>(o_attn, at);
  rank_kernel<<<BB, LS, 0, stream>>>(at, ord);
  attn_mean_kernel<<<BB, 256, 0, stream>>>(o_iat, ati);
  rank_kernel<<<BB, LS, 0, stream>>>(ati, ordi);

  ce_indices_kernel<<<BB, LS, 0, stream>>>(ord,  gi_s,  o_gs,  o_rm);
  ce_indices_kernel<<<BB, LS, 0, stream>>>(ordi, gi_si, o_gsi, o_rmi);
  copy_i2f_kernel<<<(unsigned)((GT_SZ + 255) / 256), 256, 0, stream>>>(gi_t,  o_gt,  (long long)GT_SZ);
  copy_i2f_kernel<<<(unsigned)((GT_SZ + 255) / 256), 256, 0, stream>>>(gi_ti, o_gti, (long long)GT_SZ);

  gather_kernel<<<(unsigned)((X_SZ + 255) / 256), 256, 0, stream>>>(xn,  ord,  o_x);
  gather_kernel<<<(unsigned)((X_SZ + 255) / 256), 256, 0, stream>>>(xin, ordi, o_xi);

  // ======================= stage 2: MLP + mixer2 =======================
  // LN of pruned tensors BEFORE any in-place update (xori semantics)
  ln_kernel<<<BB * LP, 256, 0, stream>>>(o_x,  n2_g, n2_b, lnA, CCH);
  ln_kernel<<<BB * LP, 256, 0, stream>>>(o_xi, n2_g, n2_b, lnB, CCH);

  // ---- x update ----
  launch_gemm(stream, lnA, fc1_w, fc1_b, big,
              BB * LP, FF, CCH, CCH, FF, FF,
              0, 0, 0, 0, 0, 0, 1, 1, 1.0f, 0, 1);
  launch_gemm(stream, big, fc2_w, fc2_b, t1,
              BB * LP, CCH, FF, FF, CCH, CCH,
              0, 0, 0, 0, 0, 0, 1, 1, 1.0f, 0, 0);
  launch_gemm(stream, lnB, ad2_dw, ad2_db, mh,
              BB * LP, ADH, CCH, CCH, ADH, ADH,
              0, 0, 0, 0, 0, 0, 1, 1, 1.0f, 0, 1);
  launch_gemm(stream, mh, ad2_uw, ad2_ub, t2,
              BB * LP, CCH, ADH, ADH, CCH, CCH,
              0, 0, 0, 0, 0, 0, 1, 1, 1.0f, 0, 0);
  addip2_kernel<<<(unsigned)((X_SZ + 255) / 256), 256, 0, stream>>>(o_x, t1, t2, (long long)X_SZ);

  // ---- xi update ----
  launch_gemm(stream, lnB, fc1_w, fc1_b, big,
              BB * LP, FF, CCH, CCH, FF, FF,
              0, 0, 0, 0, 0, 0, 1, 1, 1.0f, 0, 1);
  launch_gemm(stream, big, fc2_w, fc2_b, t1,
              BB * LP, CCH, FF, FF, CCH, CCH,
              0, 0, 0, 0, 0, 0, 1, 1, 1.0f, 0, 0);
  launch_gemm(stream, lnA, ad2_dw, ad2_db, mh,
              BB * LP, ADH, CCH, CCH, ADH, ADH,
              0, 0, 0, 0, 0, 0, 1, 1, 1.0f, 0, 1);
  launch_gemm(stream, mh, ad2_uw, ad2_ub, t2,
              BB * LP, CCH, ADH, ADH, CCH, CCH,
              0, 0, 0, 0, 0, 0, 1, 1, 1.0f, 0, 0);
  addip2_kernel<<<(unsigned)((X_SZ + 255) / 256), 256, 0, stream>>>(o_xi, t1, t2, (long long)X_SZ);
}
